// MHSAttention_15272903705138
// MI455X (gfx1250) — matrix operand order repaired and checked by reading, not compiled, not run on hardware
//
#include <hip/hip_runtime.h>

#define B_ 8
#define T_ 1024
#define E_ 1024
#define H_ 16
#define D_ 64

typedef __attribute__((ext_vector_type(16))) _Float16 v16h;
typedef __attribute__((ext_vector_type(8)))  _Float16 v8h;
typedef __attribute__((ext_vector_type(8)))  float    v8f;
typedef __attribute__((ext_vector_type(4)))  float    v4f;

static __device__ __forceinline__ v16h cmb(v8h lo, v8h hi) {
  return __builtin_shufflevector(lo, hi, 0,1,2,3,4,5,6,7,8,9,10,11,12,13,14,15);
}
static __device__ __forceinline__ v8h ldg8(const _Float16* p) { return *(const v8h*)p; }

// Row reductions across the 16-lane half-wave (wave32; masks 1,2,4,8 stay in-half).
static __device__ __forceinline__ float redmax16(float v) {
  v = fmaxf(v, __shfl_xor(v, 1, 32));
  v = fmaxf(v, __shfl_xor(v, 2, 32));
  v = fmaxf(v, __shfl_xor(v, 4, 32));
  v = fmaxf(v, __shfl_xor(v, 8, 32));
  return v;
}
static __device__ __forceinline__ float redsum16(float v) {
  v += __shfl_xor(v, 1, 32);
  v += __shfl_xor(v, 2, 32);
  v += __shfl_xor(v, 4, 32);
  v += __shfl_xor(v, 8, 32);
  return v;
}

// Kernel 1: W[m][h][e][d] f32 -> Wt[m][h][d][e] f16 (contiguous rows for WMMA B-frags).
__global__ __launch_bounds__(256) void wtrans(const float* __restrict__ wq,
                                              const float* __restrict__ wk,
                                              const float* __restrict__ wv,
                                              _Float16* __restrict__ wt) {
  const int idx = blockIdx.x;            // 3*H*D blocks
  const int m   = idx / (H_ * D_);
  const int hd  = idx % (H_ * D_);
  const int h   = hd / D_;
  const int d   = hd % D_;
  const float* w = (m == 0 ? wq : (m == 1 ? wk : wv)) + (size_t)h * E_ * D_;
  _Float16* o = wt + (((size_t)m * H_ + h) * D_ + d) * E_;
  for (int e = threadIdx.x; e < E_; e += 256)
    o[e] = (_Float16)w[(size_t)e * D_ + d];
}

// Kernel 2: QKV projection, block-cooperative.
// Block = 8 waves = 64 rows x 8 n-tiles. The 64x32 f32 x-tile is streamed into
// LDS with gfx1250 async global->LDS copies (double buffered, ASYNCcnt), and
// shared by all 8 waves; each wave runs 4 WMMAs (4 m-tiles) per k-step.
__global__ __launch_bounds__(256) void qkv_proj(const float* __restrict__ x,
                                                const _Float16* __restrict__ wt,
                                                _Float16* __restrict__ qg,
                                                _Float16* __restrict__ kg,
                                                _Float16* __restrict__ vtg) {
  // 64 rows x 32 floats, padded to 36 floats/row (144B, 16B aligned stride).
  __shared__ float __attribute__((aligned(16))) xt[2][64 * 36];
  const int tid  = threadIdx.x;
  const int wv   = tid >> 5;
  const int cb   = blockIdx.x >> 7;      // 24 column blocks
  const int rb   = blockIdx.x & 127;     // 128 row blocks (64 rows each)
  const int tg0  = rb << 6;              // first global row of this block
  const int gn   = cb * 8 + wv;          // global n-tile 0..191
  const int m    = gn >> 6;              // 0=q 1=k 2=v
  const int rest = gn & 63;
  const int h    = rest >> 2;
  const int c    = rest & 3;
  const int lane = tid & 31, ln = lane & 15, hi = lane >> 4;

  // Async-copy mapping: 512 16B segments per tile, 2 per thread.
  auto issue = [&](int buf, int kk) {
#pragma unroll
    for (int i = 0; i < 2; ++i) {
      const int s   = tid * 2 + i;
      const int row = s >> 3;            // 8 segments of 4 floats per row
      const int c4  = s & 7;
      const float* g = x + (size_t)(tg0 + row) * E_ + kk + c4 * 4;
      unsigned lo = (unsigned)(uintptr_t)&xt[buf][row * 36 + c4 * 4];
      unsigned long long ga = (unsigned long long)(uintptr_t)g;
      asm volatile("global_load_async_to_lds_b128 %0, %1, off"
                   :: "v"(lo), "v"(ga) : "memory");
    }
  };

  const _Float16* wrow = wt + (((size_t)m * H_ + h) * D_ + c * 16 + ln) * E_ + 16 * hi;
  v8f acc[4] = {};

  issue(0, 0);
  int cur = 0;
  for (int kk = 0; kk < E_; kk += 32, cur ^= 1) {
    // My share of the current tile has landed; my reads of the other buffer
    // (about to be overwritten) are complete; then meet the whole block.
    asm volatile("s_wait_asynccnt 0x0" ::: "memory");
    asm volatile("s_wait_dscnt 0x0" ::: "memory");
    __syncthreads();
    if (kk + 32 < E_) issue(cur ^ 1, kk + 32);

    // B-frag (32x16 f16): lane = column c*16+ln, k = kk+16*hi .. +16 (contiguous).
    v16h b = cmb(ldg8(wrow + kk), ldg8(wrow + kk + 8));

    const float* base = &xt[cur][0];
#pragma unroll
    for (int mt = 0; mt < 4; ++mt) {
      // A-frag (16x32 f16): lane = row mt*16+ln, k = 16*hi .. +16 (contiguous).
      const float* pr = base + (mt * 16 + ln) * 36 + 16 * hi;
      v4f f0 = *(const v4f*)(pr);
      v4f f1 = *(const v4f*)(pr + 4);
      v4f f2 = *(const v4f*)(pr + 8);
      v4f f3 = *(const v4f*)(pr + 12);
      v16h a;
#pragma unroll
      for (int i = 0; i < 4; ++i) {
        a[i]      = (_Float16)f0[i];
        a[4 + i]  = (_Float16)f1[i];
        a[8 + i]  = (_Float16)f2[i];
        a[12 + i] = (_Float16)f3[i];
      }
      acc[mt] = __builtin_amdgcn_wmma_f32_16x16x32_f16(false, a, false, b,
                                                       (short)0, acc[mt], false, false);
    }
  }

  // Store: C layout VGPR r -> row r + 8*hi, col = ln.
#pragma unroll
  for (int mt = 0; mt < 4; ++mt) {
    const int t0   = tg0 + mt * 16;
    const int bb   = t0 >> 10;           // / T_
    const int tloc = t0 & (T_ - 1);
    const int bh   = bb * H_ + h;
    const int d    = c * 16 + ln;
    if (m == 2) {
      _Float16* o = vtg + ((size_t)bh * D_ + d) * T_ + tloc;   // V^T [BH][D][T]
#pragma unroll
      for (int r = 0; r < 8; ++r) o[r + 8 * hi] = (_Float16)acc[mt][r];
    } else {
      _Float16* o = (m == 0 ? qg : kg) + ((size_t)bh * T_ + tloc) * D_ + d;
#pragma unroll
      for (int r = 0; r < 8; ++r) o[(size_t)(r + 8 * hi) * D_] = (_Float16)acc[mt][r];
    }
  }
}

// Kernel 3: flash attention. One wave per (b,h, 16-row Q tile).
__global__ __launch_bounds__(256) void attn_fa(const _Float16* __restrict__ qg,
                                               const _Float16* __restrict__ kg,
                                               const _Float16* __restrict__ vtg,
                                               float* __restrict__ out) {
  __shared__ _Float16 __attribute__((aligned(16))) pbuf[8][16 * 40];
  const int wv   = threadIdx.x >> 5;
  const int wave = blockIdx.x * 8 + wv;  // 8192 waves total
  const int bh   = wave >> 6;            // 0..127
  const int tl   = (wave & 63) << 4;     // Q row tile start
  const int b    = bh >> 4;
  const int h    = bh & 15;
  const int lane = threadIdx.x & 31;
  const int ln   = lane & 15;
  const int hi   = lane >> 4;

  // Q A-frags for d=0..31 and d=32..63 (loaded once).
  const _Float16* qrow = qg + ((size_t)bh * T_ + tl + ln) * D_;
  v16h qa0 = cmb(ldg8(qrow + 16 * hi),      ldg8(qrow + 8 + 16 * hi));
  v16h qa1 = cmb(ldg8(qrow + 32 + 16 * hi), ldg8(qrow + 40 + 16 * hi));

  v8f o0 = {}, o1 = {}, o2 = {}, o3 = {};
  float mr[8], lr[8];
#pragma unroll
  for (int r = 0; r < 8; ++r) { mr[r] = -1e30f; lr[r] = 0.f; }

  const _Float16* kb = kg  + (size_t)bh * T_ * D_;
  const _Float16* vb = vtg + (size_t)bh * D_ * T_;
  const int jend = tl + 16;              // keys needed: s <= tl+15

  for (int j0 = 0; j0 < jend; j0 += 32) {
    // K B-frags: B = K^T, lane column j0(+16)+ln = row of K (contiguous in d).
    const _Float16* kr0 = kb + (size_t)(j0 + ln) * D_ + 16 * hi;
    const _Float16* kr1 = kr0 + 16 * D_;
    __builtin_prefetch(kr0 + 32 * D_, 0, 1);
    v16h k0a = cmb(ldg8(kr0),      ldg8(kr0 + 8));
    v16h k0b = cmb(ldg8(kr0 + 32), ldg8(kr0 + 40));
    v16h k1a = cmb(ldg8(kr1),      ldg8(kr1 + 8));
    v16h k1b = cmb(ldg8(kr1 + 32), ldg8(kr1 + 40));

    v8f s0 = {}, s1 = {};
    s0 = __builtin_amdgcn_wmma_f32_16x16x32_f16(false, qa0, false, k0a, (short)0, s0, false, false);
    s0 = __builtin_amdgcn_wmma_f32_16x16x32_f16(false, qa1, false, k0b, (short)0, s0, false, false);
    s1 = __builtin_amdgcn_wmma_f32_16x16x32_f16(false, qa0, false, k1a, (short)0, s1, false, false);
    s1 = __builtin_amdgcn_wmma_f32_16x16x32_f16(false, qa1, false, k1b, (short)0, s1, false, false);

    // Online softmax over this 32-key chunk; causal mask via selects (EXEC untouched).
#pragma unroll
    for (int r = 0; r < 8; ++r) {
      const int row = tl + r + 8 * hi;
      float x0 = (row >= j0 + ln)      ? s0[r] * 0.125f : -1e30f;
      float x1 = (row >= j0 + 16 + ln) ? s1[r] * 0.125f : -1e30f;
      const float mx = redmax16(fmaxf(x0, x1));
      const float mn = fmaxf(mr[r], mx);
      const float al = __expf(mr[r] - mn);
      mr[r] = mn;
      const float p0 = __expf(x0 - mn);
      const float p1 = __expf(x1 - mn);
      lr[r] = lr[r] * al + redsum16(p0 + p1);
      o0[r] *= al; o1[r] *= al; o2[r] *= al; o3[r] *= al;
      pbuf[wv][(r + 8 * hi) * 40 + ln]      = (_Float16)p0;
      pbuf[wv][(r + 8 * hi) * 40 + 16 + ln] = (_Float16)p1;
    }
    // C-layout -> A-layout transpose through per-wave LDS; DS ops are in-order
    // per wave, the wait + memory clobber stops compiler reordering.
    asm volatile("s_wait_dscnt 0x0" ::: "memory");
    const _Float16* pr = &pbuf[wv][ln * 40 + 16 * hi];
    v16h pa = cmb(*(const v8h*)pr, *(const v8h*)(pr + 8));

    // O += P * V ; V^T rows contiguous: lane col d=c*16+ln, k = s = j0+16*hi..+16.
    const _Float16* vr = vb + (size_t)ln * T_ + j0 + 16 * hi;
    o0 = __builtin_amdgcn_wmma_f32_16x16x32_f16(false, pa, false,
         cmb(ldg8(vr),            ldg8(vr + 8)),            (short)0, o0, false, false);
    o1 = __builtin_amdgcn_wmma_f32_16x16x32_f16(false, pa, false,
         cmb(ldg8(vr + 16 * T_),  ldg8(vr + 16 * T_ + 8)),  (short)0, o1, false, false);
    o2 = __builtin_amdgcn_wmma_f32_16x16x32_f16(false, pa, false,
         cmb(ldg8(vr + 32 * T_),  ldg8(vr + 32 * T_ + 8)),  (short)0, o2, false, false);
    o3 = __builtin_amdgcn_wmma_f32_16x16x32_f16(false, pa, false,
         cmb(ldg8(vr + 48 * T_),  ldg8(vr + 48 * T_ + 8)),  (short)0, o3, false, false);
  }

  // Normalize and write out [B,T,H*D] fp32.
  float inv[8];
#pragma unroll
  for (int r = 0; r < 8; ++r) inv[r] = 1.0f / lr[r];
  float* ob = out + ((size_t)b * T_ + tl) * (H_ * D_) + h * D_;
#pragma unroll
  for (int r = 0; r < 8; ++r) {
    float* orow = ob + (size_t)(r + 8 * hi) * (H_ * D_);
    orow[ln]      = o0[r] * inv[r];
    orow[16 + ln] = o1[r] * inv[r];
    orow[32 + ln] = o2[r] * inv[r];
    orow[48 + ln] = o3[r] * inv[r];
  }
}

extern "C" void kernel_launch(void* const* d_in, const int* in_sizes, int n_in,
                              void* d_out, int out_size, void* d_ws, size_t ws_size,
                              hipStream_t stream) {
  (void)in_sizes; (void)n_in; (void)out_size; (void)ws_size;
  const float* x  = (const float*)d_in[0];
  const float* Wq = (const float*)d_in[1];
  const float* Wk = (const float*)d_in[2];
  const float* Wv = (const float*)d_in[3];
  float* out = (float*)d_out;

  // Workspace layout (f16 halves):
  //   wt : 3*H*D*E            =  3,145,728
  //   q  : B*H*T*D            =  8,388,608   [BH][T][D]
  //   k  : B*H*T*D            =  8,388,608   [BH][T][D]
  //   vt : B*H*D*T            =  8,388,608   [BH][D][T]  (transposed for PV)
  _Float16* ws = (_Float16*)d_ws;
  _Float16* wt = ws;
  _Float16* q  = wt + (size_t)3 * H_ * D_ * E_;
  _Float16* k  = q  + (size_t)B_ * H_ * T_ * D_;
  _Float16* vt = k  + (size_t)B_ * H_ * T_ * D_;

  wtrans  <<<3 * H_ * D_, 256, 0, stream>>>(Wq, Wk, Wv, wt);
  // 128 row blocks (64 rows) x 24 column blocks (8 n-tiles each) = 3072 blocks
  qkv_proj<<<3072, 256, 0, stream>>>(x, wt, q, k, vt);
  // B*H * (T/16) = 8192 waves -> 1024 blocks of 8 waves
  attn_fa <<<1024, 256, 0, stream>>>(q, k, vt, out);
}
